// CrossSectionalRanker_58746562675051
// MI455X (gfx1250) — matrix operand order repaired and checked by reading, not compiled, not run on hardware
//
#include <hip/hip_runtime.h>

#define DIN     64
#define NROWS   1000000
#define KSHORT  2048
#define NB_CNT  245          // ceil(1e6 / 4096)

typedef __attribute__((ext_vector_type(16))) __bf16 v16bf;
typedef __attribute__((ext_vector_type(8)))  float  v8f;
typedef __attribute__((ext_vector_type(4)))  int    v4i;

struct U4 { unsigned int x, y, z, w; };
union  ABf { v16bf v; unsigned int u[8]; };

// ---- optional CDNA5 async memory->LDS path (compile-safe via __has_builtin) ----
#if defined(__has_builtin)
# if __has_builtin(__builtin_amdgcn_global_load_async_to_lds_b128) && \
     __has_builtin(__builtin_amdgcn_s_wait_asynccnt)
#  define HAS_ASYNC 1
# endif
#endif
#ifndef HAS_ASYNC
# define HAS_ASYNC 0
#endif

#if HAS_ASYNC
typedef __attribute__((address_space(1))) v4i* g128p;
typedef __attribute__((address_space(3))) v4i* l128p;
__device__ __forceinline__ void async_b128(const void* gsrc, void* ldst){
  __builtin_amdgcn_global_load_async_to_lds_b128(
      (g128p)(unsigned long long)(uintptr_t)gsrc,
      (l128p)(unsigned int)(uintptr_t)ldst, 0, 0);
}
__device__ __forceinline__ void async_wait(){ __builtin_amdgcn_s_wait_asynccnt(0); }
#endif

__device__ __forceinline__ float  bf2f(__bf16 b){ return (float)b; }
__device__ __forceinline__ __bf16 f2bf(float f){ return (__bf16)f; }
__device__ __forceinline__ float  gelu_f(float x){ return 0.5f*x*(1.0f+erff(x*0.70710678118654752f)); }
__device__ __forceinline__ float  sigmoid_f(float x){ return 1.0f/(1.0f+__expf(-x)); }
__device__ __forceinline__ unsigned int fkey(float f){
  unsigned int u = __float_as_uint(f);
  return (u >> 31) ? ~u : (u | 0x80000000u);
}

// ---------------------------------------------------------------------------
// WMMA helpers.  Activations A: bf16 [16][K] row-major in LDS.
// Weights B: bf16 [N][K] row-major ("WT") in LDS.
// ---------------------------------------------------------------------------
__device__ __forceinline__ v8f wmma_tile_k(const __bf16* As, int strideA,
                                           const __bf16* Bs, int strideB,
                                           int K, v8f acc){
  const int lane = threadIdx.x & 31;
  const int half = lane >> 4, l16 = lane & 15;
  const unsigned int* Ar = (const unsigned int*)(As + l16 * strideA);
  const unsigned int* Br = (const unsigned int*)(Bs + l16 * strideB);
  for (int k0 = 0; k0 < K; k0 += 32){
    const int d = k0 >> 1;
    ABf a, b; U4 t;
    t = *(const U4*)(Ar + d + half*4);       a.u[0]=t.x; a.u[1]=t.y; a.u[2]=t.z; a.u[3]=t.w;
    t = *(const U4*)(Ar + d + 8 + half*4);   a.u[4]=t.x; a.u[5]=t.y; a.u[6]=t.z; a.u[7]=t.w;
    t = *(const U4*)(Br + d + half*4);       b.u[0]=t.x; b.u[1]=t.y; b.u[2]=t.z; b.u[3]=t.w;
    t = *(const U4*)(Br + d + 8 + half*4);   b.u[4]=t.x; b.u[5]=t.y; b.u[6]=t.z; b.u[7]=t.w;
    acc = __builtin_amdgcn_wmma_f32_16x16x32_bf16(false, a.v, false, b.v,
                                                  (short)0, acc, false, false);
  }
  return acc;
}

// C layout: VGPR r -> row = r + (lane>=16 ? 8 : 0), col = lane&15
// OP: 0 = bias only, 1 = bias+GELU, 2 = bias+sigmoid
template<int OP>
__device__ __forceinline__ void store_tile_bf16(__bf16* C, int stride, v8f c,
                                                const float* bias){
  const int lane = threadIdx.x & 31, half = lane >> 4, l16 = lane & 15;
  const float bv = bias ? bias[l16] : 0.f;     // one load per tile
  #pragma unroll
  for (int r = 0; r < 8; r++){
    float v = c[r] + bv;
    if (OP == 1) v = gelu_f(v);
    if (OP == 2) v = sigmoid_f(v);
    C[(r + half*8) * stride + l16] = f2bf(v);
  }
}

template<int OP>
__device__ __forceinline__ void wave_gemm(const __bf16* A, int K,
                                          const __bf16* W, int dout,
                                          __bf16* Cdst, const float* bias){
  const int ntiles = dout >> 4;
  #pragma unroll
  for (int nt = 0; nt < 8; nt++){
    if (nt < ntiles){
      v8f acc = {0,0,0,0,0,0,0,0};
      acc = wmma_tile_k(A, K, W + nt*16*K, K, K, acc);
      store_tile_bf16<OP>(Cdst + nt*16, dout, acc, bias ? bias + nt*16 : nullptr);
    }
  }
}

// LayerNorm(+optional GELU) on one wave's 16x[dim] bf16 LDS tile
__device__ __forceinline__ void ln_act_wave(__bf16* buf, int dim,
                                            const float* g, const float* be,
                                            int dogelu, float* red){
  const int lane = threadIdx.x & 31, half = lane >> 4, r = lane & 15;
  const int h = dim >> 1;
  __bf16* row = buf + r*dim + half*h;
  float s = 0.f, ss = 0.f;
  for (int c = 0; c < h; c++){ float x = bf2f(row[c]); s += x; ss += x*x; }
  red[half*16 + r] = s; red[32 + half*16 + r] = ss;
  __syncthreads();
  float S = red[r] + red[16+r], SS = red[32+r] + red[48+r];
  float mean = S / (float)dim;
  float var  = SS / (float)dim - mean*mean;
  float inv  = rsqrtf(var + 1e-5f);
  const float* gg = g + half*h; const float* bb = be + half*h;
  for (int c = 0; c < h; c++){
    float x = (bf2f(row[c]) - mean) * inv * gg[c] + bb[c];
    if (dogelu) x = gelu_f(x);
    row[c] = f2bf(x);
  }
  __syncthreads();
}

__device__ __forceinline__ void stage_w(const __bf16* src, __bf16* dst, int n){
#if HAS_ASYNC
  for (int i = threadIdx.x; i < (n >> 3); i += blockDim.x)
    async_b128(src + i*8, dst + i*8);
  async_wait();
#else
  const unsigned int* s = (const unsigned int*)src;
  unsigned int* d = (unsigned int*)dst;
  for (int i = threadIdx.x; i < (n >> 1); i += blockDim.x) d[i] = s[i];
#endif
}
__device__ __forceinline__ void stage_f(const float* src, float* dst, int n){
  for (int i = threadIdx.x; i < n; i += blockDim.x) dst[i] = src[i];
}

// ---------------------------------------------------------------------------
// Fused backbone: stem (64->128->64, LN+GELU) + 2 gated residual blocks.
// Block = 128 threads = 4 waves; 64 rows per block, 16 rows per wave.
// ---------------------------------------------------------------------------
struct BBArgs {
  const float* features;
  const int*   gidx;
  const __bf16 *wt_s0, *wt_s1, *wt_w1[2], *wt_w2[2], *wt_wg[2];
  const float *b_s0, *g_s0, *be_s0, *b_s1, *g_s1, *be_s1;
  const float *ln_g[2], *ln_b[2], *b1[2], *b2[2], *bg[2];
  const float *base_w, *base_b, *lin_w;
  float* outBase;
  __bf16* outEnc;
};

__global__ __launch_bounds__(128) void backbone_kernel(BBArgs A){
  __shared__ __align__(16) __bf16 sW[128*64];
  __shared__ float sB[128], sG[128], sE[128];
  __shared__ __align__(16) __bf16 sX[4][16*64];
  __shared__ __align__(16) __bf16 sT1[4][16*128];
  __shared__ __align__(16) __bf16 sT2[4][16*64];
  __shared__ __align__(16) __bf16 sR[4][16*64];
  __shared__ float sRed[4][64];
  __shared__ float sLin[4][32];

  const int tid = threadIdx.x, w = tid >> 5, lane = tid & 31;
  const int rowbase = blockIdx.x * 64 + w * 16;

  // ---- load features (bf16) + fused linear-head partial dot ----
  {
    const int r = lane >> 1, hc = (lane & 1) * 32;
    const int srow = rowbase + r;
    const int grow = A.gidx ? A.gidx[srow] : srow;
    const float* fp = A.features + (size_t)grow * DIN + hc;
    float lin = 0.f;
    for (int c = 0; c < 32; c++){
      float v = fp[c];
      sX[w][r*64 + hc + c] = f2bf(v);
      if (A.outBase) lin += v * A.lin_w[hc + c];
    }
    sLin[w][lane] = lin;
  }
  __syncthreads();

  // ---- stem 0: 64 -> 128, LN+GELU ----
  stage_w(A.wt_s0, sW, 128*64);
  stage_f(A.b_s0, sB, 128); stage_f(A.g_s0, sG, 128); stage_f(A.be_s0, sE, 128);
  __syncthreads();
  wave_gemm<0>(sX[w], 64, sW, 128, sT1[w], sB);
  __syncthreads();
  ln_act_wave(sT1[w], 128, sG, sE, 1, sRed[w]);

  // ---- stem 1: 128 -> 64, LN+GELU ----
  stage_w(A.wt_s1, sW, 64*128);
  stage_f(A.b_s1, sB, 64); stage_f(A.g_s1, sG, 64); stage_f(A.be_s1, sE, 64);
  __syncthreads();
  wave_gemm<0>(sT1[w], 128, sW, 64, sX[w], sB);
  __syncthreads();
  ln_act_wave(sX[w], 64, sG, sE, 1, sRed[w]);

  // ---- residual blocks ----
  for (int b = 0; b < 2; b++){
    for (int i = lane; i < 16*64; i += 32){ sR[w][i] = sX[w][i]; sT2[w][i] = sX[w][i]; }
    stage_f(A.ln_g[b], sG, 64); stage_f(A.ln_b[b], sE, 64);
    __syncthreads();
    ln_act_wave(sT2[w], 64, sG, sE, 0, sRed[w]);          // normalized copy
    stage_w(A.wt_w1[b], sW, 128*64); stage_f(A.b1[b], sB, 128);
    __syncthreads();
    wave_gemm<1>(sT2[w], 64, sW, 128, sT1[w], sB);        // GELU(h@w1+b1)
    __syncthreads();
    stage_w(A.wt_w2[b], sW, 64*128); stage_f(A.b2[b], sB, 64);
    __syncthreads();
    wave_gemm<0>(sT1[w], 128, sW, 64, sT2[w], sB);        // h
    __syncthreads();
    stage_w(A.wt_wg[b], sW, 64*64); stage_f(A.bg[b], sB, 64);
    __syncthreads();
    wave_gemm<2>(sR[w], 64, sW, 64, sT1[w], sB);          // gate (stride-64 region)
    __syncthreads();
    for (int i = lane; i < 16*64; i += 32){
      float x = bf2f(sR[w][i]) + bf2f(sT1[w][i]) * bf2f(sT2[w][i]);
      sX[w][i] = f2bf(x);
    }
    __syncthreads();
  }

  if (A.outEnc){
    __bf16* dst = A.outEnc + (size_t)rowbase * 64;
    for (int i = lane; i < 16*64; i += 32) dst[i] = sX[w][i];
  }
  if (A.outBase){
    const int half = lane >> 4, r = lane & 15;
    const __bf16* row = sX[w] + r*64 + half*32;
    float d = 0.f;
    for (int c = 0; c < 32; c++) d += bf2f(row[c]) * A.base_w[half*32 + c];
    sRed[w][half*16 + r] = d;
    __syncthreads();
    if (half == 0){
      float dot = sRed[w][r] + sRed[w][16 + r] + A.base_b[0];
      float lin = sLin[w][2*r] + sLin[w][2*r + 1];
      A.outBase[rowbase + r] = 0.5f * lin + 0.5f * dot;
    }
  }
}

// ---------------------------------------------------------------------------
// Generic K-chunked WMMA GEMM: C[M,N] = A[M,K] @ WT[N,K]^T  (+bias, epilogue)
// MODE: 0 plain, 1 GELU, 2 qkv-scatter, 3 add residual.
// Grid: (M/64, ceil(N/256)), 128 threads.
// ---------------------------------------------------------------------------
struct GArgs {
  const __bf16* A; int lda;
  const __bf16* WT; int K, N, M;
  const float* bias; float scale;
  const float* resid; int ldr;
  float* outF; __bf16* outB; int ldo, colOff;
  __bf16 *qp, *kp, *vT;
};

template<int MODE>
__global__ __launch_bounds__(128) void gemm_kernel(GArgs g){
  __shared__ __align__(16) __bf16 sW[256*64];
  __shared__ __align__(16) __bf16 sA[4][16*64];
  __shared__ float sB[256];
  const int tid = threadIdx.x, w = tid >> 5, lane = tid & 31;
  const int mbase = blockIdx.x * 64 + w * 16;
  const int nbase = blockIdx.y * 256;
  const int NT = (g.N - nbase < 256) ? (g.N - nbase) : 256;
  const int ntiles = NT >> 4;

  if (g.bias) for (int i = tid; i < NT; i += 128) sB[i] = g.bias[nbase + i];

  v8f acc[16];
  #pragma unroll
  for (int i = 0; i < 16; i++) acc[i] = (v8f){0,0,0,0,0,0,0,0};

  for (int k0 = 0; k0 < g.K; k0 += 64){
    const int Kc  = (g.K - k0 < 64) ? (g.K - k0) : 64;
    const int dwc = Kc >> 1;
    __syncthreads();
#if HAS_ASYNC
    if (dwc == 32){
      // WT chunk: NT rows x 8 16-byte packets
      for (int i = tid; i < NT*8; i += 128){
        int n = i >> 3, seg = i & 7;
        async_b128(g.WT + (size_t)(nbase + n)*g.K + k0 + seg*8,
                   (char*)sW + n*128 + seg*16);
      }
      // A chunk per wave: 16 rows x 8 packets
      for (int i = lane; i < 16*8; i += 32){
        int rr = i >> 3, seg = i & 7;
        async_b128(g.A + (size_t)(mbase + rr)*g.lda + k0 + seg*8,
                   (char*)sA[w] + rr*128 + seg*16);
      }
      async_wait();
    } else
#endif
    {
      for (int i = tid; i < NT*32; i += 128){
        int n = i >> 5, dk = i & 31;
        unsigned int v = 0;
        if (dk < dwc)
          v = ((const unsigned int*)(g.WT + (size_t)(nbase + n)*g.K + k0))[dk];
        ((unsigned int*)sW)[n*32 + dk] = v;
      }
      for (int i = lane; i < 16*32; i += 32){
        int rr = i >> 5, dk = i & 31;
        unsigned int v = 0;
        if (dk < dwc)
          v = ((const unsigned int*)(g.A + (size_t)(mbase + rr)*g.lda + k0))[dk];
        ((unsigned int*)sA[w])[rr*32 + dk] = v;
      }
    }
    if (k0 + 64 < g.K)
      __builtin_prefetch(g.WT + (size_t)(nbase + (tid & 63))*g.K + k0 + 64, 0, 1);
    __syncthreads();
    #pragma unroll
    for (int nt = 0; nt < 16; nt++)
      if (nt < ntiles)
        acc[nt] = wmma_tile_k(sA[w], 64, sW + nt*16*64, 64, 64, acc[nt]);
  }

  const int half = lane >> 4, l16 = lane & 15;
  #pragma unroll
  for (int nt = 0; nt < 16; nt++){
    if (nt >= ntiles) continue;
    const float bv = g.bias ? sB[nt*16 + l16] : 0.f;
    #pragma unroll
    for (int r2 = 0; r2 < 8; r2++){
      const int row = mbase + r2 + half*8;
      const int col = nbase + nt*16 + l16;
      float v = acc[nt][r2] * g.scale + bv;
      if (MODE == 1) v = gelu_f(v);
      if (MODE == 3) v += g.resid[(size_t)row * g.ldr + col];
      if (MODE == 2){
        if (col < 64)      { int h=col>>4,  d=col&15;        g.qp[((size_t)h*KSHORT+row)*32 + d] = f2bf(v); }
        else if (col < 128){ int c=col-64,  h=c>>4, d=c&15;  g.kp[((size_t)h*KSHORT+row)*32 + d] = f2bf(v); }
        else               { int c=col-128, h=c>>4, d=c&15;  g.vT[((size_t)h*16 + d)*KSHORT + row] = f2bf(v); }
      } else {
        if (g.outF) g.outF[(size_t)row * g.ldo + col + g.colOff] = v;
        if (g.outB) g.outB[(size_t)row * g.ldo + col + g.colOff] = f2bf(v);
      }
    }
  }
}

// ---------------------------------------------------------------------------
// Row softmax over 2048 cols, f32 in -> bf16 probs out. One block per row.
// ---------------------------------------------------------------------------
__global__ __launch_bounds__(256) void softmax_kernel(const float* S, __bf16* P){
  const int row = blockIdx.x, tid = threadIdx.x;
  const float* s = S + (size_t)row * KSHORT;
  __bf16* p = P + (size_t)row * KSHORT;
  __shared__ float red[8];
  float m = -1e30f;
  for (int i = tid; i < KSHORT; i += 256) m = fmaxf(m, s[i]);
  for (int o = 16; o > 0; o >>= 1) m = fmaxf(m, __shfl_xor(m, o));
  if ((tid & 31) == 0) red[tid >> 5] = m;
  __syncthreads();
  float M = red[0];
  for (int i = 1; i < 8; i++) M = fmaxf(M, red[i]);
  __syncthreads();
  float sum = 0.f;
  for (int i = tid; i < KSHORT; i += 256) sum += __expf(s[i] - M);
  for (int o = 16; o > 0; o >>= 1) sum += __shfl_xor(sum, o);
  if ((tid & 31) == 0) red[tid >> 5] = sum;
  __syncthreads();
  float SUM = 0.f;
  for (int i = 0; i < 8; i++) SUM += red[i];
  float inv = 1.f / SUM;
  for (int i = tid; i < KSHORT; i += 256) p[i] = f2bf(__expf(s[i] - M) * inv);
}

// LayerNorm over dim=64 rows: one wave per row, 8 rows per block
__global__ __launch_bounds__(256) void ln_rows_kernel(const float* X, const float* g,
                                                      const float* b, float* outF,
                                                      __bf16* outB){
  const int w = threadIdx.x >> 5, lane = threadIdx.x & 31;
  const int row = blockIdx.x * 8 + w;
  const float* x = X + (size_t)row * 64;
  float a0 = x[lane], a1 = x[lane + 32];
  float s = a0 + a1, ss = a0*a0 + a1*a1;
  for (int o = 16; o > 0; o >>= 1){ s += __shfl_xor(s, o); ss += __shfl_xor(ss, o); }
  float mean = s * (1.f/64.f), var = ss * (1.f/64.f) - mean*mean;
  float inv = rsqrtf(var + 1e-5f);
  float y0 = (a0 - mean)*inv*g[lane] + b[lane];
  float y1 = (a1 - mean)*inv*g[lane+32] + b[lane+32];
  outF[(size_t)row*64 + lane] = y0;      outF[(size_t)row*64 + lane + 32] = y1;
  outB[(size_t)row*64 + lane] = f2bf(y0); outB[(size_t)row*64 + lane + 32] = f2bf(y1);
}

// --------------------------- top-k (radix select) ---------------------------
__global__ void fill_u32_kernel(unsigned int* p, unsigned int v, int n){
  int i = blockIdx.x * blockDim.x + threadIdx.x;
  if (i < n) p[i] = v;
}
__global__ void hist0_kernel(const float* s, int n, unsigned int* h){
  for (int i = blockIdx.x*blockDim.x + threadIdx.x; i < n; i += gridDim.x*blockDim.x)
    atomicAdd(&h[fkey(s[i]) >> 20], 1u);
}
__global__ void hist1_kernel(const float* s, int n, const unsigned int* sel, unsigned int* h){
  unsigned b0 = sel[0];
  for (int i = blockIdx.x*blockDim.x + threadIdx.x; i < n; i += gridDim.x*blockDim.x){
    unsigned k = fkey(s[i]);
    if ((k >> 20) == b0) atomicAdd(&h[(k >> 8) & 0xFFFu], 1u);
  }
}
__global__ void hist2_kernel(const float* s, int n, const unsigned int* sel, unsigned int* h){
  unsigned pre = (sel[0] << 12) | sel[2];
  for (int i = blockIdx.x*blockDim.x + threadIdx.x; i < n; i += gridDim.x*blockDim.x){
    unsigned k = fkey(s[i]);
    if ((k >> 8) == pre) atomicAdd(&h[k & 0xFFu], 1u);
  }
}
__global__ void sel_kernel(const unsigned int* hist, int bins, unsigned int* sel, int stage){
  if (threadIdx.x != 0 || blockIdx.x != 0) return;
  unsigned need = (stage == 0) ? (unsigned)KSHORT : sel[stage*2 - 1];
  unsigned cum = 0; int b = bins - 1;
  for (; b > 0; b--){ unsigned c = hist[b]; if (cum + c >= need) break; cum += c; }
  sel[stage*2] = (unsigned)b; sel[stage*2 + 1] = need - cum;
  if (stage == 2) sel[6] = (sel[0] << 20) | (sel[2] << 8) | sel[4];
}
__global__ __launch_bounds__(256) void cnt_kernel(const float* s, int n,
                                                  const unsigned int* sel,
                                                  unsigned int* bcGt, unsigned int* bcEq){
  const unsigned T = sel[6];
  __shared__ unsigned sg[8], se[8];
  const int tid = threadIdx.x;
  size_t base = (size_t)blockIdx.x * 4096 + (size_t)tid * 16;
  unsigned cg = 0, ce = 0;
  for (int j = 0; j < 16; j++){
    size_t i = base + j;
    if (i < (size_t)n){ unsigned k = fkey(s[i]); cg += (k > T); ce += (k == T); }
  }
  for (int o = 16; o > 0; o >>= 1){ cg += __shfl_xor(cg, o); ce += __shfl_xor(ce, o); }
  if ((tid & 31) == 0){ sg[tid >> 5] = cg; se[tid >> 5] = ce; }
  __syncthreads();
  if (tid == 0){
    unsigned a = 0, b = 0;
    for (int i = 0; i < 8; i++){ a += sg[i]; b += se[i]; }
    bcGt[blockIdx.x] = a; bcEq[blockIdx.x] = b;
  }
}
__global__ void scan_kernel(unsigned int* bcGt, unsigned int* bcEq, int nb, unsigned int* sel){
  if (threadIdx.x != 0 || blockIdx.x != 0) return;
  unsigned ag = 0, ae = 0;
  for (int i = 0; i < nb; i++){
    unsigned t = bcGt[i]; bcGt[i] = ag; ag += t;
    unsigned e = bcEq[i]; bcEq[i] = ae; ae += e;
  }
  sel[7] = ag;   // G = count strictly greater
}
__global__ __launch_bounds__(256) void scatter_kernel(const float* s, int n,
                                                      const unsigned int* sel,
                                                      const unsigned int* bcGt,
                                                      const unsigned int* bcEq,
                                                      int* idxList){
  const unsigned T = sel[6], G = sel[7], needEq = sel[5];
  __shared__ unsigned pg[256], pe[256];
  const int tid = threadIdx.x;
  size_t base = (size_t)blockIdx.x * 4096 + (size_t)tid * 16;
  unsigned cg = 0, ce = 0;
  for (int j = 0; j < 16; j++){
    size_t i = base + j;
    if (i < (size_t)n){ unsigned k = fkey(s[i]); cg += (k > T); ce += (k == T); }
  }
  pg[tid] = cg; pe[tid] = ce;
  __syncthreads();
  if (tid == 0){
    unsigned a = 0, e = 0;
    for (int i = 0; i < 256; i++){
      unsigned t = pg[i]; pg[i] = a; a += t;
      unsigned q = pe[i]; pe[i] = e; e += q;
    }
  }
  __syncthreads();
  unsigned og = bcGt[blockIdx.x] + pg[tid];
  unsigned oe = bcEq[blockIdx.x] + pe[tid];
  for (int j = 0; j < 16; j++){
    size_t i = base + j;
    if (i < (size_t)n){
      unsigned k = fkey(s[i]);
      if (k > T) idxList[og++] = (int)i;
      else if (k == T){ if (oe < needEq) idxList[G + oe] = (int)i; oe++; }
    }
  }
}

// weight convert: f32 [din][dout] -> bf16 transposed [dout][din]
__global__ void convw_kernel(const float* W, __bf16* WT, int din, int dout){
  int n = din * dout;
  for (int i = blockIdx.x*blockDim.x + threadIdx.x; i < n; i += gridDim.x*blockDim.x){
    int k = i / dout, nn = i % dout;
    WT[(size_t)nn * din + k] = f2bf(W[i]);
  }
}

// rerank head + merge + scatter into d_out
__global__ void merge_kernel(const float* t, const float* rh_w, const float* rh_b,
                             const int* idx, float* out){
  int i = blockIdx.x * blockDim.x + threadIdx.x;
  if (i >= KSHORT) return;
  const float* r = t + (size_t)i * 64;
  float d = rh_b[0];
  for (int c = 0; c < 64; c++) d += r[c] * rh_w[c];
  int gi = idx[i];
  out[gi] = 0.5f * out[gi] + 0.5f * d;
}

// ---------------------------------------------------------------------------
extern "C" void kernel_launch(void* const* d_in, const int* in_sizes, int n_in,
                              void* d_out, int out_size, void* d_ws, size_t ws_size,
                              hipStream_t stream){
  (void)in_sizes; (void)n_in; (void)out_size; (void)ws_size;
  const float* F = (const float*)d_in[0];
  float* base = (float*)d_out;

  // ---- carve workspace ----
  char* wsp = (char*)d_ws; size_t off = 0;
  auto carve = [&](size_t bytes)->char*{
    off = (off + 255) & ~(size_t)255; char* p = wsp + off; off += bytes; return p;
  };
  __bf16* wt_s0   = (__bf16*)carve(128*64*2);
  __bf16* wt_s1   = (__bf16*)carve(64*128*2);
  __bf16* wt_w1[2]; __bf16* wt_w2[2]; __bf16* wt_wg[2];
  for (int b = 0; b < 2; b++){
    wt_w1[b] = (__bf16*)carve(128*64*2);
    wt_w2[b] = (__bf16*)carve(64*128*2);
    wt_wg[b] = (__bf16*)carve(64*64*2);
  }
  __bf16* wt_proj = (__bf16*)carve(64*64*2);
  __bf16* wt_qkv[2]; __bf16* wt_wo[2]; __bf16* wt_fw1[2]; __bf16* wt_fw2[2];
  for (int l = 0; l < 2; l++){
    wt_qkv[l] = (__bf16*)carve(192*64*2);
    wt_wo[l]  = (__bf16*)carve(64*64*2);
    wt_fw1[l] = (__bf16*)carve(256*64*2);
    wt_fw2[l] = (__bf16*)carve(64*256*2);
  }
  unsigned int* hist0 = (unsigned int*)carve(4096*4);
  unsigned int* hist1 = (unsigned int*)carve(4096*4);
  unsigned int* hist2 = (unsigned int*)carve(256*4);
  unsigned int* sel   = (unsigned int*)carve(16*4);
  unsigned int* bcGt  = (unsigned int*)carve(256*4);
  unsigned int* bcEq  = (unsigned int*)carve(256*4);
  int* idxList        = (int*)carve(KSHORT*4);
  __bf16* enc   = (__bf16*)carve((size_t)KSHORT*64*2);
  float*  tF    = (float*) carve((size_t)KSHORT*64*4);
  __bf16* tB    = (__bf16*)carve((size_t)KSHORT*64*2);
  __bf16* qp    = (__bf16*)carve((size_t)4*KSHORT*32*2);
  __bf16* kp    = (__bf16*)carve((size_t)4*KSHORT*32*2);
  __bf16* vT    = (__bf16*)carve((size_t)4*16*KSHORT*2);
  float*  scores= (float*) carve((size_t)KSHORT*KSHORT*4);
  __bf16* probs = (__bf16*)carve((size_t)KSHORT*KSHORT*2);
  __bf16* oB    = (__bf16*)carve((size_t)KSHORT*64*2);
  float*  tsum  = (float*) carve((size_t)KSHORT*64*4);
  __bf16* ff1B  = (__bf16*)carve((size_t)KSHORT*256*2);

  // ---- convert weights to bf16 transposed ----
  auto conv = [&](int i, int din, int dout, __bf16* dst){
    int n = din * dout;
    convw_kernel<<<(n + 255)/256, 256, 0, stream>>>((const float*)d_in[i], dst, din, dout);
  };
  conv(1, 64, 128, wt_s0);
  conv(5, 128, 64, wt_s1);
  for (int b = 0; b < 2; b++){
    int o = 9 + b*8;
    conv(o+2, 64, 128, wt_w1[b]);
    conv(o+4, 128, 64, wt_w2[b]);
    conv(o+6, 64, 64,  wt_wg[b]);
  }
  conv(28, 64, 64, wt_proj);
  for (int l = 0; l < 2; l++){
    int o = 30 + l*12;
    conv(o,   64, 192, wt_qkv[l]);
    conv(o+2, 64, 64,  wt_wo[l]);
    conv(o+6, 64, 256, wt_fw1[l]);
    conv(o+8, 256, 64, wt_fw2[l]);
  }

  // ---- zero histograms and q/k pad buffers ----
  auto fill0 = [&](unsigned int* p, int n){
    fill_u32_kernel<<<(n + 255)/256, 256, 0, stream>>>(p, 0u, n);
  };
  fill0(hist0, 4096); fill0(hist1, 4096); fill0(hist2, 256); fill0(sel, 16);
  fill0((unsigned int*)qp, (int)((size_t)4*KSHORT*32*2/4));
  fill0((unsigned int*)kp, (int)((size_t)4*KSHORT*32*2/4));

  // ---- phase 1: backbone over all rows -> base scores ----
  BBArgs ba{};
  ba.features = F; ba.gidx = nullptr;
  ba.wt_s0 = wt_s0; ba.wt_s1 = wt_s1;
  ba.b_s0 = (const float*)d_in[2]; ba.g_s0 = (const float*)d_in[3]; ba.be_s0 = (const float*)d_in[4];
  ba.b_s1 = (const float*)d_in[6]; ba.g_s1 = (const float*)d_in[7]; ba.be_s1 = (const float*)d_in[8];
  for (int b = 0; b < 2; b++){
    int o = 9 + b*8;
    ba.wt_w1[b] = wt_w1[b]; ba.wt_w2[b] = wt_w2[b]; ba.wt_wg[b] = wt_wg[b];
    ba.ln_g[b] = (const float*)d_in[o];   ba.ln_b[b] = (const float*)d_in[o+1];
    ba.b1[b]   = (const float*)d_in[o+3]; ba.b2[b]   = (const float*)d_in[o+5];
    ba.bg[b]   = (const float*)d_in[o+7];
  }
  ba.base_w = (const float*)d_in[25]; ba.base_b = (const float*)d_in[26];
  ba.lin_w  = (const float*)d_in[27];
  ba.outBase = base; ba.outEnc = nullptr;
  backbone_kernel<<<NROWS/64, 128, 0, stream>>>(ba);

  // ---- phase 2: deterministic top-2048 radix select + compaction ----
  hist0_kernel<<<1024, 256, 0, stream>>>(base, NROWS, hist0);
  sel_kernel<<<1, 32, 0, stream>>>(hist0, 4096, sel, 0);
  hist1_kernel<<<1024, 256, 0, stream>>>(base, NROWS, sel, hist1);
  sel_kernel<<<1, 32, 0, stream>>>(hist1, 4096, sel, 1);
  hist2_kernel<<<1024, 256, 0, stream>>>(base, NROWS, sel, hist2);
  sel_kernel<<<1, 32, 0, stream>>>(hist2, 256, sel, 2);
  cnt_kernel<<<NB_CNT, 256, 0, stream>>>(base, NROWS, sel, bcGt, bcEq);
  scan_kernel<<<1, 32, 0, stream>>>(bcGt, bcEq, NB_CNT, sel);
  scatter_kernel<<<NB_CNT, 256, 0, stream>>>(base, NROWS, sel, bcGt, bcEq, idxList);

  // ---- phase 3: recompute backbone for shortlist -> encoded bf16 ----
  BBArgs bg2 = ba;
  bg2.gidx = idxList; bg2.outBase = nullptr; bg2.outEnc = enc;
  backbone_kernel<<<KSHORT/64, 128, 0, stream>>>(bg2);

  // ---- generic GEMM dispatch ----
  auto gemm = [&](const __bf16* A, int lda, const __bf16* WT, int K, int N, int M,
                  const float* bias, float scale, int mode,
                  const float* resid, int ldr,
                  float* outF, __bf16* outB, int ldo, int colOff){
    GArgs g{};
    g.A = A; g.lda = lda; g.WT = WT; g.K = K; g.N = N; g.M = M;
    g.bias = bias; g.scale = scale; g.resid = resid; g.ldr = ldr;
    g.outF = outF; g.outB = outB; g.ldo = ldo; g.colOff = colOff;
    g.qp = qp; g.kp = kp; g.vT = vT;
    dim3 grid(M/64, (N + 255)/256);
    switch (mode){
      case 0: gemm_kernel<0><<<grid, 128, 0, stream>>>(g); break;
      case 1: gemm_kernel<1><<<grid, 128, 0, stream>>>(g); break;
      case 2: gemm_kernel<2><<<grid, 128, 0, stream>>>(g); break;
      default: gemm_kernel<3><<<grid, 128, 0, stream>>>(g); break;
    }
  };

  // ---- proj: t = enc @ proj_w + proj_b ----
  gemm(enc, 64, wt_proj, 64, 64, KSHORT, (const float*)d_in[29], 1.f, 0,
       nullptr, 0, tF, tB, 64, 0);

  // ---- phase 4: 2 transformer layers ----
  for (int l = 0; l < 2; l++){
    int o = 30 + l*12;
    const float* bqkv = (const float*)d_in[o+1];
    const float* bo   = (const float*)d_in[o+3];
    const float* l1g  = (const float*)d_in[o+4];
    const float* l1b  = (const float*)d_in[o+5];
    const float* fb1  = (const float*)d_in[o+7];
    const float* fb2  = (const float*)d_in[o+9];
    const float* l2g  = (const float*)d_in[o+10];
    const float* l2b  = (const float*)d_in[o+11];

    // qkv with scatter into padded q/k and transposed v
    gemm(tB, 64, wt_qkv[l], 64, 192, KSHORT, bqkv, 1.f, 2, nullptr, 0,
         nullptr, nullptr, 0, 0);
    for (int h = 0; h < 4; h++){
      // scores = (q @ k^T) * 1/sqrt(16)
      gemm(qp + (size_t)h*KSHORT*32, 32, kp + (size_t)h*KSHORT*32, 32, KSHORT, KSHORT,
           nullptr, 0.25f, 0, nullptr, 0, scores, nullptr, KSHORT, 0);
      softmax_kernel<<<KSHORT, 256, 0, stream>>>(scores, probs);
      // o_h = probs @ v  -> columns [h*16, h*16+16) of oB
      gemm(probs, KSHORT, vT + (size_t)h*16*KSHORT, KSHORT, 16, KSHORT,
           nullptr, 1.f, 0, nullptr, 0, nullptr, oB, 64, h*16);
    }
    // wo + residual, then LN1
    gemm(oB, 64, wt_wo[l], 64, 64, KSHORT, bo, 1.f, 3, tF, 64, tsum, nullptr, 64, 0);
    ln_rows_kernel<<<KSHORT/8, 256, 0, stream>>>(tsum, l1g, l1b, tF, tB);
    // FFN
    gemm(tB, 64, wt_fw1[l], 64, 256, KSHORT, fb1, 1.f, 1, nullptr, 0,
         nullptr, ff1B, 256, 0);
    gemm(ff1B, 256, wt_fw2[l], 256, 64, KSHORT, fb2, 1.f, 3, tF, 64, tsum, nullptr, 64, 0);
    ln_rows_kernel<<<KSHORT/8, 256, 0, stream>>>(tsum, l2g, l2b, tF, tB);
  }

  // ---- rerank head + merge into d_out ----
  merge_kernel<<<KSHORT/256, 256, 0, stream>>>(tF, (const float*)d_in[54],
                                               (const float*)d_in[55], idxList, base);
}
